// SelfAttention_32770600469142
// MI455X (gfx1250) — compile-verified, matrix operand order checked by reading, not run on hardware
//
#include <hip/hip_runtime.h>
#include <hip/hip_bf16.h>

// ---------------------------------------------------------------------------
// Self-attention for MI455X (gfx1250): all matmuls via V_WMMA_F32_16X16X32_BF16
//   kernel 1: QKV projection GEMM, double-buffered LDS staging (register-
//             staged global_load_b128 -> v_cvt_pk_bf16_f32 -> ds_store) so
//             loads for K-slab k+32 overlap the 8 WMMAs of slab k.
//             Bias folded, Q pre-scaled by 1/sqrt(64); K stored row-major
//             (S,D), V stored transposed (D,S) for direct B-fragment loads.
//   kernel 2: per (b,h,64-query-tile) attention: scores in ~194KB dynamic LDS
//             (CDNA5 WGP has 320KB), masked softmax (bias row staged in LDS),
//             P.V via WMMA against V^T.
// ---------------------------------------------------------------------------

typedef __bf16 bf16;
typedef __attribute__((ext_vector_type(8)))  __bf16 v8bf;
typedef __attribute__((ext_vector_type(16))) __bf16 v16bf;
typedef __attribute__((ext_vector_type(8)))  float  v8f;

#define EMBED   768
#define HIDDEN  768
#define HEADS   12
#define HEAD_DIM 64
#define BATCH   16
#define SEQ     512
#define M_TOTAL (BATCH*SEQ)   // 8192

// Load one 16x32 bf16 WMMA operand fragment from a row-major tile (stride ld,
// in elements; rows 16B aligned). ISA 7.12.2 "16-bit A 16x32":
//   lanes 0..15  (row = lane)    hold K = 0..7  and 16..23
//   lanes 16..31 (row = lane-16) hold K = 8..15 and 24..31
// Serves B operands too when the tile is the transposed B (NxK row-major).
static __device__ __forceinline__ v16bf load_frag_bf16(const bf16* __restrict__ tile, int ld) {
  const int lane = threadIdx.x & 31;
  const int hi16 = lane >> 4;
  const int row  = lane & 15;
  const bf16* rp = tile + row * ld;
  const int kb0 = hi16 ? 8  : 0;
  const int kb1 = hi16 ? 24 : 16;
  v8bf lo = *(const v8bf*)(rp + kb0);
  v8bf hi = *(const v8bf*)(rp + kb1);
  return __builtin_shufflevector(lo, hi, 0,1,2,3,4,5,6,7,8,9,10,11,12,13,14,15);
}

static __device__ __forceinline__ v8f wmma_bf16(v16bf a, v16bf b, v8f c) {
  // 8 args: (neg_a, A, neg_b, B, c_mod, C, reuse_a, reuse_b)
  return __builtin_amdgcn_wmma_f32_16x16x32_bf16(false, a, false, b, (short)0, c, false, false);
}

// ---------------------------------------------------------------------------
// Kernel 1: C(8192x768) = X(8192x768) * W(768x768) + bias  (z selects Q/K/V)
// 128x128 workgroup tile, 8 waves (4x2), each wave: 2x4 16x16 accumulators.
// Double-buffered LDS, register-staged global loads.
// ---------------------------------------------------------------------------
__global__ __launch_bounds__(256) void qkv_proj_kernel(
    const float* __restrict__ X,
    const float* __restrict__ Wq, const float* __restrict__ Wk, const float* __restrict__ Wv,
    const float* __restrict__ bq, const float* __restrict__ bk, const float* __restrict__ bv,
    bf16* __restrict__ Qo, bf16* __restrict__ Ko, bf16* __restrict__ Vt)
{
  __shared__ bf16 Xs[2][128*40];   // 128 rows x 32 K (stride 40 => 16B-aligned)
  __shared__ bf16 Ws[2][128*40];   // transposed W tile: Ws[n][k] = W[kc+k][n0+n]

  const int m0 = blockIdx.x * 128;
  const int n0 = blockIdx.y * 128;
  const int z  = blockIdx.z;
  const float* W    = (z == 0) ? Wq : (z == 1) ? Wk : Wv;
  const float* bias = (z == 0) ? bq : (z == 1) ? bk : bv;

  const int t    = threadIdx.x;
  const int lane = t & 31;
  const int wave = t >> 5;
  const int wr   = wave >> 1;   // 0..3 : 32-row group
  const int wc   = wave & 1;    // 0..1 : 64-col group

  // Register staging for the next K-slab (8 x global_load_b128 per thread-pass)
  float4 fx[4], fw[4];

  auto load_tiles = [&](int kc) {
    #pragma unroll
    for (int it = 0; it < 4; ++it) {
      const int idx = it * 1024 + t * 4;
      const int row = idx >> 5, col = idx & 31;
      fx[it] = *(const float4*)(X + (size_t)(m0 + row) * EMBED + kc + col);
    }
    #pragma unroll
    for (int it = 0; it < 4; ++it) {
      const int idx = it * 1024 + t * 4;
      const int k = idx >> 7, n = idx & 127;
      fw[it] = *(const float4*)(W + (size_t)(kc + k) * HIDDEN + n0 + n);
    }
  };
  auto store_tiles = [&](int buf) {
    #pragma unroll
    for (int it = 0; it < 4; ++it) {
      const int idx = it * 1024 + t * 4;
      const int row = idx >> 5, col = idx & 31;
      bf16* p = &Xs[buf][row * 40 + col];
      p[0] = (bf16)fx[it].x; p[1] = (bf16)fx[it].y;
      p[2] = (bf16)fx[it].z; p[3] = (bf16)fx[it].w;
    }
    #pragma unroll
    for (int it = 0; it < 4; ++it) {
      const int idx = it * 1024 + t * 4;
      const int k = idx >> 7, n = idx & 127;
      Ws[buf][(n + 0) * 40 + k] = (bf16)fw[it].x;
      Ws[buf][(n + 1) * 40 + k] = (bf16)fw[it].y;
      Ws[buf][(n + 2) * 40 + k] = (bf16)fw[it].z;
      Ws[buf][(n + 3) * 40 + k] = (bf16)fw[it].w;
    }
  };

  v8f acc[2][4];
  #pragma unroll
  for (int i = 0; i < 2; ++i)
    #pragma unroll
    for (int j = 0; j < 4; ++j) acc[i][j] = (v8f){};

  // Prologue: fill buffer 0.
  load_tiles(0);
  store_tiles(0);
  __syncthreads();

  int cur = 0;
  for (int kc = 0; kc < EMBED; kc += 32) {
    const bool has_next = (kc + 32) < EMBED;   // uniform across the block
    if (has_next) load_tiles(kc + 32);         // global loads in flight now

    v16bf a[2], bb[4];
    #pragma unroll
    for (int i = 0; i < 2; ++i) a[i]  = load_frag_bf16(&Xs[cur][(wr * 32 + i * 16) * 40], 40);
    #pragma unroll
    for (int j = 0; j < 4; ++j) bb[j] = load_frag_bf16(&Ws[cur][(wc * 64 + j * 16) * 40], 40);
    #pragma unroll
    for (int i = 0; i < 2; ++i)
      #pragma unroll
      for (int j = 0; j < 4; ++j)
        acc[i][j] = wmma_bf16(a[i], bb[j], acc[i][j]);

    if (has_next) store_tiles(cur ^ 1);        // lands behind the WMMAs
    __syncthreads();
    cur ^= 1;
  }

  // Write back per C-layout: VGPR q -> M = q + 8*(lane>=16), N = lane&15.
  // z-branch hoisted out of the inner loops (uniform per workgroup).
  const int hi16 = lane >> 4, r = lane & 15;
  if (z == 0) {
    #pragma unroll
    for (int i = 0; i < 2; ++i)
      #pragma unroll
      for (int j = 0; j < 4; ++j) {
        const int n = n0 + wc * 64 + j * 16 + r;
        const float bv_ = bias[n];
        const int h = n >> 6, d = n & 63;
        #pragma unroll
        for (int q = 0; q < 8; ++q) {
          const int m = m0 + wr * 32 + i * 16 + q + 8 * hi16;
          const int b = m >> 9, s = m & 511;
          Qo[((size_t)(b * HEADS + h) * SEQ + s) * HEAD_DIM + d] =
              (bf16)((acc[i][j][q] + bv_) * 0.125f);
        }
      }
  } else if (z == 1) {
    #pragma unroll
    for (int i = 0; i < 2; ++i)
      #pragma unroll
      for (int j = 0; j < 4; ++j) {
        const int n = n0 + wc * 64 + j * 16 + r;
        const float bv_ = bias[n];
        const int h = n >> 6, d = n & 63;
        #pragma unroll
        for (int q = 0; q < 8; ++q) {
          const int m = m0 + wr * 32 + i * 16 + q + 8 * hi16;
          const int b = m >> 9, s = m & 511;
          Ko[((size_t)(b * HEADS + h) * SEQ + s) * HEAD_DIM + d] = (bf16)(acc[i][j][q] + bv_);
        }
      }
  } else {
    #pragma unroll
    for (int i = 0; i < 2; ++i)
      #pragma unroll
      for (int j = 0; j < 4; ++j) {
        const int n = n0 + wc * 64 + j * 16 + r;
        const float bv_ = bias[n];
        const int h = n >> 6, d = n & 63;
        #pragma unroll
        for (int q = 0; q < 8; ++q) {
          const int m = m0 + wr * 32 + i * 16 + q + 8 * hi16;
          const int b = m >> 9, s = m & 511;
          Vt[((size_t)(b * HEADS + h) * HEAD_DIM + d) * SEQ + s] = (bf16)(acc[i][j][q] + bv_);
        }
      }
  }
}

// ---------------------------------------------------------------------------
// Kernel 2: attention for one (b, h, 64-query tile). 8 waves.
//   scores: waves (mt = w&3 query sub-tile, kh = w>>2 key half) -> LDS fp32
//   softmax: 4 lanes per row, shfl_xor reductions, bias row staged in LDS
//   out:    waves (mt, d-half), K=512 WMMA loop over bf16 probs in LDS
// Dynamic LDS: 64*512*4 (scores) + 64*512*2 (bf16 probs) + 512*4 (bias)
//            = 198656 B  (CDNA5 WGP LDS is 320KB).
// ---------------------------------------------------------------------------
__global__ __launch_bounds__(256) void attention_kernel(
    const bf16* __restrict__ Qg, const bf16* __restrict__ Kg, const bf16* __restrict__ Vt,
    const float* __restrict__ mask, float* __restrict__ out)
{
  extern __shared__ char smem[];
  float* sc     = (float*)smem;                        // [64][512] fp32 scores
  bf16*  pb     = (bf16*)(smem + 64 * 512 * 4);        // [64][512] bf16 probs
  float* bias_s = (float*)(smem + 64 * 512 * 4 + 64 * 512 * 2);  // [512]

  const int q0 = blockIdx.x * 64;
  const int h  = blockIdx.y;
  const int b  = blockIdx.z;
  const int t    = threadIdx.x;
  const int lane = t & 31;
  const int wave = t >> 5;
  const int hi16 = lane >> 4, r = lane & 15;

  const bf16* qbase = Qg + ((size_t)(b * HEADS + h) * SEQ + q0) * HEAD_DIM;
  const bf16* kbase = Kg + (size_t)(b * HEADS + h) * SEQ * HEAD_DIM;
  const bf16* vbase = Vt + (size_t)(b * HEADS + h) * HEAD_DIM * SEQ;

  // Stage additive key-mask bias into LDS once (visible after the next barrier).
  if (t < 128) {
    const float4 mv = *(const float4*)(mask + (size_t)b * SEQ + t * 4);
    bias_s[t * 4 + 0] = mv.x * -1.0e9f;
    bias_s[t * 4 + 1] = mv.y * -1.0e9f;
    bias_s[t * 4 + 2] = mv.z * -1.0e9f;
    bias_s[t * 4 + 3] = mv.w * -1.0e9f;
  }

  // ---- scores = (Q/8 incl. bias) . K^T : 64 x 512 ----
  {
    const int mt = wave & 3;     // 16-query sub-tile
    const int kh = wave >> 2;    // key half: 256 keys each
    v16bf a0 = load_frag_bf16(qbase + (mt * 16) * HEAD_DIM + 0,  HEAD_DIM);
    v16bf a1 = load_frag_bf16(qbase + (mt * 16) * HEAD_DIM + 32, HEAD_DIM);
    #pragma unroll 4
    for (int nt = 0; nt < 16; ++nt) {
      const int key0 = kh * 256 + nt * 16;
      v16bf vb0 = load_frag_bf16(kbase + (size_t)key0 * HEAD_DIM + 0,  HEAD_DIM);
      v16bf vb1 = load_frag_bf16(kbase + (size_t)key0 * HEAD_DIM + 32, HEAD_DIM);
      v8f c = (v8f){};
      c = wmma_bf16(a0, vb0, c);
      c = wmma_bf16(a1, vb1, c);
      #pragma unroll
      for (int q = 0; q < 8; ++q)
        sc[(mt * 16 + q + 8 * hi16) * 512 + key0 + r] = c[q];
    }
  }
  __syncthreads();

  // ---- masked softmax over each of the 64 rows (4 lanes per row) ----
  {
    const int row  = t >> 2;
    const int part = t & 3;
    const int k0   = part * 128;
    float* rowp = sc + row * 512;
    float mx = -3.0e38f;
    for (int k = 0; k < 128; ++k) {
      const int kk = k0 + k;
      const float v = rowp[kk] + bias_s[kk];
      rowp[kk] = v;
      mx = fmaxf(mx, v);
    }
    mx = fmaxf(mx, __shfl_xor(mx, 1, 4));
    mx = fmaxf(mx, __shfl_xor(mx, 2, 4));
    float sum = 0.0f;
    for (int k = 0; k < 128; ++k) {
      const float p = __expf(rowp[k0 + k] - mx);
      rowp[k0 + k] = p;
      sum += p;
    }
    sum += __shfl_xor(sum, 1, 4);
    sum += __shfl_xor(sum, 2, 4);
    const float inv = 1.0f / sum;
    for (int k = 0; k < 128; ++k)
      pb[row * 512 + k0 + k] = (bf16)(rowp[k0 + k] * inv);
  }
  __syncthreads();

  // ---- out = P(64x512) . V(512x64), V supplied transposed (D x S) ----
  {
    const int mt = wave & 3;     // query sub-tile
    const int nh = wave >> 2;    // d half: 32 dims each
    v8f acc[2] = { (v8f){}, (v8f){} };
    #pragma unroll 4
    for (int kk = 0; kk < 512; kk += 32) {
      v16bf a = load_frag_bf16(pb + (mt * 16) * 512 + kk, 512);
      #pragma unroll
      for (int nt = 0; nt < 2; ++nt) {
        v16bf vb = load_frag_bf16(vbase + (size_t)(nh * 32 + nt * 16) * SEQ + kk, SEQ);
        acc[nt] = wmma_bf16(a, vb, acc[nt]);
      }
    }
    #pragma unroll
    for (int nt = 0; nt < 2; ++nt) {
      const int d   = nh * 32 + nt * 16 + r;
      const int col = h * HEAD_DIM + d;
      #pragma unroll
      for (int q = 0; q < 8; ++q) {
        const int s = q0 + mt * 16 + q + 8 * hi16;
        out[((size_t)b * SEQ + s) * HIDDEN + col] = acc[nt][q];
      }
    }
  }
}

// ---------------------------------------------------------------------------
extern "C" void kernel_launch(void* const* d_in, const int* in_sizes, int n_in,
                              void* d_out, int out_size, void* d_ws, size_t ws_size,
                              hipStream_t stream) {
  (void)in_sizes; (void)n_in; (void)out_size; (void)ws_size;
  const float* X    = (const float*)d_in[0];
  const float* mask = (const float*)d_in[1];
  const float* Wq   = (const float*)d_in[2];
  const float* bq   = (const float*)d_in[3];
  const float* Wk   = (const float*)d_in[4];
  const float* bk   = (const float*)d_in[5];
  const float* Wv   = (const float*)d_in[6];
  const float* bv   = (const float*)d_in[7];
  float* out = (float*)d_out;

  // Workspace: Q (B,H,S,D) bf16, K (B,H,S,D) bf16, V^T (B,H,D,S) bf16
  const size_t QK = (size_t)BATCH * HEADS * SEQ * HEAD_DIM;  // elements each
  bf16* q  = (bf16*)d_ws;
  bf16* k  = q + QK;
  bf16* vt = k + QK;

  dim3 g1(M_TOTAL / 128, HIDDEN / 128, 3);
  qkv_proj_kernel<<<g1, 256, 0, stream>>>(X, Wq, Wk, Wv, bq, bk, bv, q, k, vt);

  dim3 g2(SEQ / 64, HEADS, BATCH);
  attention_kernel<<<g2, 256, 198656, stream>>>(q, k, vt, mask, out);
}
